// FlaxGPTNeoSelfAttention_86431921864934
// MI455X (gfx1250) — matrix operand order repaired and checked by reading, not compiled, not run on hardware
//
#include <hip/hip_runtime.h>

// ---------------------------------------------------------------------------
// CDNA5 (gfx1250) GPT-Neo self-attention: bf16 WMMA GEMMs + flash attention.
// All WMMA fragments are fed by 16-byte LDS/global loads.
// ---------------------------------------------------------------------------

typedef __attribute__((ext_vector_type(16))) __bf16 v16bf;
typedef __attribute__((ext_vector_type(8)))  float  v8f;
typedef int v4i_t __attribute__((ext_vector_type(4)));

#define BATCH   2
#define SEQ     2048
#define DMODEL  2048
#define NHEAD   16
#define HDIM    128

// Optional gfx1250 async global->LDS path (guarded; sync fallback otherwise)
#if defined(__has_builtin)
#if __has_builtin(__builtin_amdgcn_global_load_async_to_lds_b128)
#define USE_ASYNC_LDS 1
#endif
#endif
#ifndef USE_ASYNC_LDS
#define USE_ASYNC_LDS 0
#endif

#if USE_ASYNC_LDS
typedef __attribute__((address_space(1))) v4i_t* gptr_v4i;
typedef __attribute__((address_space(3))) v4i_t* lptr_v4i;
static __device__ __forceinline__ void async_b128(const void* g, void* l) {
    __builtin_amdgcn_global_load_async_to_lds_b128((gptr_v4i)g, (lptr_v4i)l, 0, 0);
}
static __device__ __forceinline__ void wait_async0() {
#if __has_builtin(__builtin_amdgcn_s_wait_asynccnt)
    __builtin_amdgcn_s_wait_asynccnt(0);
#else
    asm volatile("s_wait_asynccnt 0x0" ::: "memory");
#endif
}
#endif

static __device__ __forceinline__ unsigned short f2bf(float f) {
    union { float f; unsigned u; } x; x.f = f;
    unsigned r = x.u + 0x7FFFu + ((x.u >> 16) & 1u);   // round-to-nearest-even
    return (unsigned short)(r >> 16);
}

union AFrag { v16bf v; uint4 u128[2]; unsigned u32[8]; unsigned short u16[16]; };
union CFrag { v8f   v; float f[8]; };

static __device__ __forceinline__ v8f wmma_bf16(v16bf a, v16bf b, v8f c) {
    return __builtin_amdgcn_wmma_f32_16x16x32_bf16(false, a, false, b, (short)0, c, false, false);
}

// ---------------------------------------------------------------------------
// Kernel 1a: fp32 -> bf16 cast (hidden states)
// ---------------------------------------------------------------------------
__global__ __launch_bounds__(256) void cast_f32_bf16(
    const float* __restrict__ src, unsigned short* __restrict__ dst, int n)
{
    int i = (blockIdx.x * 256 + threadIdx.x) * 4;
    if (i + 3 < n) {
        float4 f = *(const float4*)(src + i);
        union { unsigned short s[4]; uint2 v; } o;
        o.s[0] = f2bf(f.x); o.s[1] = f2bf(f.y);
        o.s[2] = f2bf(f.z); o.s[3] = f2bf(f.w);
        *(uint2*)(dst + i) = o.v;
    } else {
        for (; i < n; i++) dst[i] = f2bf(src[i]);
    }
}

// ---------------------------------------------------------------------------
// Kernel 1b: fp32 -> bf16 cast + transpose: dst[c][r] = src[r][c]
// 32x32 tiles through LDS, fully coalesced both sides.
// ---------------------------------------------------------------------------
__global__ __launch_bounds__(256) void transpose_cast_f32_bf16(
    const float* __restrict__ src, unsigned short* __restrict__ dst,
    int rows, int cols)
{
    __shared__ unsigned short tile[32][36];
    const int t  = threadIdx.x;
    const int r0 = blockIdx.y * 32, c0 = blockIdx.x * 32;
    const int lr = t >> 3;            // 0..31
    const int lc = (t & 7) << 2;      // 0,4,...,28

    float4 f = *(const float4*)&src[(size_t)(r0 + lr) * cols + c0 + lc];
    union { unsigned short s[4]; uint2 v; } p;
    p.s[0] = f2bf(f.x); p.s[1] = f2bf(f.y); p.s[2] = f2bf(f.z); p.s[3] = f2bf(f.w);
    *(uint2*)&tile[lr][lc] = p.v;
    __syncthreads();

    union { unsigned short s[4]; uint2 v; } o;
    o.s[0] = tile[lc + 0][lr]; o.s[1] = tile[lc + 1][lr];
    o.s[2] = tile[lc + 2][lr]; o.s[3] = tile[lc + 3][lr];
    *(uint2*)&dst[(size_t)(c0 + lr) * rows + r0 + lc] = o.v;
}

// ---------------------------------------------------------------------------
// Kernel 2: C[M,N] = A[M,K] x Wt[N,K]^T   (A row-major, W pre-transposed)
// Block: 8 waves. C tile: 32(M) x 128(N); wave w owns 16 cols.
// Every fragment = 2 x ds_load_b128. Output: f32+bias | bf16 | bf16-transposed.
// ---------------------------------------------------------------------------
__global__ __launch_bounds__(256) void gemm_bf16_tn(
    const unsigned short* __restrict__ A,  const unsigned short* __restrict__ Wt,
    unsigned short* __restrict__ outB, float* __restrict__ outF,
    unsigned short* __restrict__ outT, const float* __restrict__ bias,
    int M, int N, int K)
{
    __shared__ __align__(16) unsigned short As[32][40];   // stride 80B (16B-aligned)
    __shared__ __align__(16) unsigned short Bs[128][40];

    const int t    = threadIdx.x;
    const int lane = t & 31;
    const int wave = t >> 5;
    const int lcol = lane & 15;
    const int lhi  = lane >> 4;
    const int m0   = blockIdx.y * 32;
    const int nblk = blockIdx.x * 128;

    CFrag c0, c1;
#pragma unroll
    for (int r = 0; r < 8; r++) { c0.f[r] = 0.f; c1.f[r] = 0.f; }

    const int arow = t >> 3, acol = (t & 7) << 2;       // A staging map
    const int brow = t >> 1, bcol = (t & 1) << 4;       // B staging map

    for (int k0 = 0; k0 < K; k0 += 32) {
        __syncthreads();
        // --- stage A (32x32) and B (128x32) tiles ---
        *(uint2*)&As[arow][acol] =
            *(const uint2*)&A[(size_t)(m0 + arow) * K + k0 + acol];
        {
            const uint4* g = (const uint4*)&Wt[(size_t)(nblk + brow) * K + k0 + bcol];
            uint4 w0 = g[0], w1 = g[1];
            *(uint4*)&Bs[brow][bcol]     = w0;
            *(uint4*)&Bs[brow][bcol + 8] = w1;
        }
        if (k0 + 32 < K) {   // prefetch next K-chunk (global_prefetch_b8)
            __builtin_prefetch(&A[(size_t)(m0 + arow) * K + k0 + 32 + acol], 0, 0);
            __builtin_prefetch(&Wt[(size_t)(nblk + brow) * K + k0 + 32 + bcol], 0, 0);
        }
        __syncthreads();

        // --- fragments: A and B both interleaved-by-8 layout ---
        AFrag a0, a1, bw;
        a0.u128[0] = *(const uint4*)&As[lcol]     [     (lhi << 3)];
        a0.u128[1] = *(const uint4*)&As[lcol]     [16 + (lhi << 3)];
        a1.u128[0] = *(const uint4*)&As[16 + lcol][     (lhi << 3)];
        a1.u128[1] = *(const uint4*)&As[16 + lcol][16 + (lhi << 3)];
        bw.u128[0] = *(const uint4*)&Bs[(wave << 4) + lcol][     (lhi << 3)];
        bw.u128[1] = *(const uint4*)&Bs[(wave << 4) + lcol][16 + (lhi << 3)];

        c0.v = wmma_bf16(a0.v, bw.v, c0.v);
        c1.v = wmma_bf16(a1.v, bw.v, c1.v);
    }

    const int n = nblk + (wave << 4) + lcol;
    if (outF) {
        float bb = bias ? bias[n] : 0.f;
#pragma unroll
        for (int r = 0; r < 8; r++) {
            outF[(size_t)(m0 +      (lhi << 3) + r) * N + n] = c0.f[r] + bb;
            outF[(size_t)(m0 + 16 + (lhi << 3) + r) * N + n] = c1.f[r] + bb;
        }
    } else if (outT) {   // transposed bf16: outT[n][m], packed 16B stores
        union { unsigned short s[8]; uint4 v; } p0, p1;
#pragma unroll
        for (int r = 0; r < 8; r++) { p0.s[r] = f2bf(c0.f[r]); p1.s[r] = f2bf(c1.f[r]); }
        *(uint4*)&outT[(size_t)n * M + m0 +      (lhi << 3)] = p0.v;
        *(uint4*)&outT[(size_t)n * M + m0 + 16 + (lhi << 3)] = p1.v;
    } else {
#pragma unroll
        for (int r = 0; r < 8; r++) {
            outB[(size_t)(m0 +      (lhi << 3) + r) * N + n] = f2bf(c0.f[r]);
            outB[(size_t)(m0 + 16 + (lhi << 3) + r) * N + n] = f2bf(c1.f[r]);
        }
    }
}

// ---------------------------------------------------------------------------
// Kernel 3: flash attention. Grid: (S/128, B*H). Block: 256 thr = 8 waves.
// K row-major [token][d]; V pre-transposed [d][token].
// ---------------------------------------------------------------------------
__global__ __launch_bounds__(256) void flash_attn(
    const unsigned short* __restrict__ Q, const unsigned short* __restrict__ Kb,
    const unsigned short* __restrict__ Vt, const int* __restrict__ amask,
    unsigned short* __restrict__ Ob)
{
    __shared__ __align__(16) unsigned short Ks[32][136];   // [key][d]
    __shared__ __align__(16) unsigned short Vs[128][40];   // [d][key]
    __shared__ __align__(16) unsigned short Ps[8][16][40]; // per-wave P staging

    const int t     = threadIdx.x;
    const int lane  = t & 31;
    const int wave  = t >> 5;
    const int lcol  = lane & 15;
    const int lhi   = lane >> 4;
    const int bh    = blockIdx.y;
    const int b     = bh >> 4;
    const int h     = bh & 15;
    const int q0    = blockIdx.x * 128;
    const int qbase = q0 + (wave << 4);
    const size_t headoff = (size_t)h * HDIM;
    const float  scale   = 0.08838834764831845f;  // 1/sqrt(128)
    const int    Mtok    = BATCH * SEQ;           // Vt row stride

    // ---- Q fragments: 2 x global b128 per d-chunk ----
    AFrag qf[4];
    {
        const unsigned short* qrow =
            Q + (size_t)(b * SEQ + qbase + lcol) * DMODEL + headoff;
#pragma unroll
        for (int c = 0; c < 4; c++) {
            qf[c].u128[0] = *(const uint4*)&qrow[c * 32 + (lhi << 4)];
            qf[c].u128[1] = *(const uint4*)&qrow[c * 32 + (lhi << 4) + 8];
        }
    }

    CFrag o[8];
#pragma unroll
    for (int d = 0; d < 8; d++)
#pragma unroll
        for (int r = 0; r < 8; r++) o[d].f[r] = 0.f;

    float m_i[8], l_i[8];
#pragma unroll
    for (int r = 0; r < 8; r++) { m_i[r] = -1e30f; l_i[r] = 0.f; }

    const int krow = t >> 3, kcol = (t & 7) << 4;   // K staging map (32x128)
    const int vrow = t >> 1, vcol = (t & 1) << 4;   // V staging map (128x32)

    const int nkt = (q0 + 128) / 32;   // causal: keys <= q0+127
    for (int kt = 0; kt < nkt; kt++) {
        const int k0 = kt * 32;
        __syncthreads();
        {
            size_t kb = (size_t)(b * SEQ + k0 + krow) * DMODEL + headoff + kcol;
            size_t vb = (size_t)(headoff + vrow) * Mtok + b * SEQ + k0 + vcol;
#if USE_ASYNC_LDS
            async_b128(&Kb[kb],     &Ks[krow][kcol]);
            async_b128(&Kb[kb + 8], &Ks[krow][kcol + 8]);
            async_b128(&Vt[vb],     &Vs[vrow][vcol]);
            async_b128(&Vt[vb + 8], &Vs[vrow][vcol + 8]);
            wait_async0();
#else
            const uint4* gk = (const uint4*)&Kb[kb];
            const uint4* gv = (const uint4*)&Vt[vb];
            uint4 ka = gk[0], kc2 = gk[1];
            uint4 va = gv[0], vc2 = gv[1];
            *(uint4*)&Ks[krow][kcol]     = ka;  *(uint4*)&Ks[krow][kcol + 8] = kc2;
            *(uint4*)&Vs[vrow][vcol]     = va;  *(uint4*)&Vs[vrow][vcol + 8] = vc2;
#endif
        }
        __syncthreads();

        if (k0 > qbase + 15) continue;   // tile fully masked for this wave

        // ---- scores: S[16q x 32k], two 16x16 column halves ----
        CFrag s0, s1;
#pragma unroll
        for (int r = 0; r < 8; r++) { s0.f[r] = 0.f; s1.f[r] = 0.f; }
#pragma unroll
        for (int c = 0; c < 4; c++) {
            AFrag bk0, bk1;   // B[d][k]: lane = key col, elements = contiguous d
            bk0.u128[0] = *(const uint4*)&Ks[lcol]     [c * 32 + (lhi << 4)];
            bk0.u128[1] = *(const uint4*)&Ks[lcol]     [c * 32 + (lhi << 4) + 8];
            bk1.u128[0] = *(const uint4*)&Ks[16 + lcol][c * 32 + (lhi << 4)];
            bk1.u128[1] = *(const uint4*)&Ks[16 + lcol][c * 32 + (lhi << 4) + 8];
            s0.v = wmma_bf16(qf[c].v, bk0.v, s0.v);
            s1.v = wmma_bf16(qf[c].v, bk1.v, s1.v);
        }

        // ---- mask + online softmax (rows live in 16-lane halves) ----
        const int kc0  = k0 + lcol;
        const int kc1  = k0 + 16 + lcol;
        const int pad0 = amask[b * SEQ + kc0] > 0;
        const int pad1 = amask[b * SEQ + kc1] > 0;
        float alpha[8];
#pragma unroll
        for (int r = 0; r < 8; r++) {
            const int q = qbase + (lhi << 3) + r;
            float v0 = (pad0 && kc0 <= q) ? s0.f[r] * scale : -1e30f;
            float v1 = (pad1 && kc1 <= q) ? s1.f[r] * scale : -1e30f;
            float mx = fmaxf(v0, v1);
#pragma unroll
            for (int off = 8; off; off >>= 1)
                mx = fmaxf(mx, __shfl_xor(mx, off, 32));
            float mn = fmaxf(m_i[r], mx);
            alpha[r] = __expf(m_i[r] - mn);
            float p0 = __expf(v0 - mn);
            float p1 = __expf(v1 - mn);
            s0.f[r] = p0; s1.f[r] = p1;
            float rs = p0 + p1;
#pragma unroll
            for (int off = 8; off; off >>= 1)
                rs += __shfl_xor(rs, off, 32);
            l_i[r] = l_i[r] * alpha[r] + rs;
            m_i[r] = mn;
        }

        // ---- rescale O accumulators ----
#pragma unroll
        for (int d = 0; d < 8; d++)
#pragma unroll
            for (int r = 0; r < 8; r++) o[d].f[r] *= alpha[r];

        // ---- P (C layout) -> LDS -> A layout fragment ----
#pragma unroll
        for (int r = 0; r < 8; r++) {
            Ps[wave][(lhi << 3) + r][lcol]      = f2bf(s0.f[r]);
            Ps[wave][(lhi << 3) + r][16 + lcol] = f2bf(s1.f[r]);
        }
        AFrag pf;
        pf.u128[0] = *(const uint4*)&Ps[wave][lcol][(lhi << 4)];
        pf.u128[1] = *(const uint4*)&Ps[wave][lcol][(lhi << 4) + 8];

        // ---- O += P x V (8 d-chunks), B from transposed V ----
#pragma unroll
        for (int d = 0; d < 8; d++) {
            AFrag bv;   // B[k][d]: lane = d col, elements = contiguous k
            bv.u128[0] = *(const uint4*)&Vs[(d << 4) + lcol][(lhi << 4)];
            bv.u128[1] = *(const uint4*)&Vs[(d << 4) + lcol][(lhi << 4) + 8];
            o[d].v = wmma_bf16(pf.v, bv.v, o[d].v);
        }
    }

    // ---- normalize, store context bf16 [B,S,D] ----
    float rl[8];
#pragma unroll
    for (int r = 0; r < 8; r++) rl[r] = 1.f / l_i[r];
#pragma unroll
    for (int d = 0; d < 8; d++) {
#pragma unroll
        for (int r = 0; r < 8; r++) {
            const int q = qbase + (lhi << 3) + r;
            Ob[(size_t)(b * SEQ + q) * DMODEL + headoff + (d << 4) + lcol] =
                f2bf(o[d].f[r] * rl[r]);
        }
    }
}

// ---------------------------------------------------------------------------
// Host-side launch
// ---------------------------------------------------------------------------
extern "C" void kernel_launch(void* const* d_in, const int* in_sizes, int n_in,
                              void* d_out, int out_size, void* d_ws, size_t ws_size,
                              hipStream_t stream) {
    (void)in_sizes; (void)n_in; (void)out_size; (void)ws_size;

    const float* hs = (const float*)d_in[0];
    const int*   am = (const int*)d_in[1];
    const float* wq = (const float*)d_in[2];
    const float* wk = (const float*)d_in[3];
    const float* wv = (const float*)d_in[4];
    const float* wo = (const float*)d_in[5];
    const float* bo = (const float*)d_in[6];
    float* out = (float*)d_out;

    const size_t nHS = (size_t)BATCH * SEQ * DMODEL;   // 8,388,608
    const size_t nW  = (size_t)DMODEL * DMODEL;        // 4,194,304

    unsigned short* hsb = (unsigned short*)d_ws;
    unsigned short* wqT = hsb + nHS;
    unsigned short* wkT = wqT + nW;
    unsigned short* wvT = wkT + nW;
    unsigned short* woT = wvT + nW;
    unsigned short* Qb  = woT + nW;
    unsigned short* Kbf = Qb + nHS;
    unsigned short* Vt  = Kbf + nHS;
    unsigned short* Cb  = Vt + nHS;   // total ~112 MB workspace

    // 1) casts (weights transposed: Wt[n][k])
    cast_f32_bf16<<<dim3((unsigned)(nHS / 1024)), 256, 0, stream>>>(hs, hsb, (int)nHS);
    dim3 gt(DMODEL / 32, DMODEL / 32);
    transpose_cast_f32_bf16<<<gt, 256, 0, stream>>>(wq, wqT, DMODEL, DMODEL);
    transpose_cast_f32_bf16<<<gt, 256, 0, stream>>>(wk, wkT, DMODEL, DMODEL);
    transpose_cast_f32_bf16<<<gt, 256, 0, stream>>>(wv, wvT, DMODEL, DMODEL);
    transpose_cast_f32_bf16<<<gt, 256, 0, stream>>>(wo, woT, DMODEL, DMODEL);

    // 2) projections: Q,K row-major bf16; V transposed bf16 [d][token]
    const int M = BATCH * SEQ;
    dim3 gg(DMODEL / 128, M / 32);
    gemm_bf16_tn<<<gg, 256, 0, stream>>>(hsb, wqT, Qb, nullptr, nullptr, nullptr, M, DMODEL, DMODEL);
    gemm_bf16_tn<<<gg, 256, 0, stream>>>(hsb, wkT, Kbf, nullptr, nullptr, nullptr, M, DMODEL, DMODEL);
    gemm_bf16_tn<<<gg, 256, 0, stream>>>(hsb, wvT, nullptr, nullptr, Vt, nullptr, M, DMODEL, DMODEL);

    // 3) flash attention
    dim3 ga(SEQ / 128, BATCH * NHEAD);
    flash_attn<<<ga, 256, 0, stream>>>(Qb, Kbf, Vt, am, Cb);

    // 4) output projection (fp32 + bias)
    gemm_bf16_tn<<<gg, 256, 0, stream>>>(Cb, woT, nullptr, out, nullptr, bo, M, DMODEL, DMODEL);
}
